// LSTMFPN_25812753449758
// MI455X (gfx1250) — matrix operand order repaired and checked by reading, not compiled, not run on hardware
//
#include <hip/hip_runtime.h>
#include <hip/hip_bf16.h>

typedef __attribute__((ext_vector_type(8)))  _Float16 v8h;
typedef __attribute__((ext_vector_type(16))) _Float16 v16h;
typedef __attribute__((ext_vector_type(8)))  float    v8f;
typedef __attribute__((ext_vector_type(4)))  int      v4i;

// ---------------------------------------------------------------------------
// Implicit-GEMM conv, NHWC f16 activations, weights packed [co][tap][ci] f16.
//
// Workgroup = WPB waves (4 or 8).  All waves share one 16-pixel M-tile; each
// wave owns a distinct 16-wide cout tile.  Per (tap, source) the block stages
// the 16 x C activation chunk into LDS once (coalesced b128 loads, padding
// zeroed at stage time), then every wave builds its A fragments from LDS
// (ds_load_b128, bank-conflict-free via +8-half row padding) while streaming
// B fragments from L2-resident packed weights.  This cuts A-side global/L2
// traffic by WPB x vs. one-wave-per-tile.
//
// Fragment layouts used by this kernel (wave32):
//   A (16x32 f16): row M = lane&15; khalf=lane>>4 holds K = [cb+16*khalf,+16)
//                  contiguous, same K order as B
//   B (32x16 f16): col N = lane&15; K = [cb+16*khalf, +16) contiguous
//   C/D (16x16 f32): VGPR r -> [M = r + 8*khalf][N = lane&15]
//
// MODE: 0 = BN+ReLU (q0..q3 = g,b,m,v), 1 = bias (q0).
// FINAL: also write f32 NCHW into out32 (network outputs).
// All H/W/C are powers of two -> shifts/masks only.
// ---------------------------------------------------------------------------
template <int KS, int MODE, int FINAL>
__global__ __launch_bounds__(256)
void wmma_conv_lds(const _Float16* __restrict__ s1, int C1,
                   const _Float16* __restrict__ s2, int C2,
                   const _Float16* __restrict__ wpk,
                   const float* __restrict__ q0, const float* __restrict__ q1,
                   const float* __restrict__ q2, const float* __restrict__ q3,
                   _Float16* __restrict__ out16, float* __restrict__ out32,
                   int Hin, int Win, int Hout, int Wout, int Cout, int stride)
{
    constexpr int PAD  = 8;                 // halves; staggers LDS bank groups
    constexpr int CMAX = 512;               // largest single staged source
    __shared__ _Float16 As[16 * (CMAX + PAD)];

    const int lane  = threadIdx.x;          // 0..31
    const int wv    = threadIdx.y;          // 0..WPB-1
    const int WPB   = blockDim.y;
    const int tid   = wv * 32 + lane;
    const int nth   = WPB * 32;
    const int col   = lane & 15;
    const int khalf = lane >> 4;

    const int l2W = 31 - __clz(Wout);
    const int l2H = 31 - __clz(Hout);

    const int co = (blockIdx.y * WPB + wv) * 16 + col;
    const int Ct = C1 + C2;
    const _Float16* wrow = wpk + (size_t)co * (KS * KS) * Ct;

    v8f acc = {};

#pragma unroll
    for (int kh = 0; kh < KS; ++kh) {
#pragma unroll
        for (int kw = 0; kw < KS; ++kw) {
            const int tap = kh * KS + kw;
            const _Float16* bptap = wrow + (size_t)tap * Ct;
            const int nsrc = s2 ? 2 : 1;                 // uniform
            for (int si = 0; si < nsrc; ++si) {
                const _Float16* src = si ? s2 : s1;
                const int C = si ? C2 : C1;
                const _Float16* bp = si ? (bptap + C1) : bptap;
                const int l2C  = 31 - __clz(C);
                const int rows = C + PAD;

                __syncthreads();   // previous chunk fully consumed
                // ---- stage 16 pixels x C channels into LDS (v8h pieces) ----
                for (int p = tid; p < 2 * C; p += nth) {     // (16*C)/8 pieces
                    const int e    = p * 8;
                    const int pixl = e >> l2C;
                    const int ch   = e & (C - 1);
                    const int pmg  = blockIdx.x * 16 + pixl;
                    const int owp  = pmg & (Wout - 1);
                    const int tq   = pmg >> l2W;
                    const int ohp  = tq & (Hout - 1);
                    const int nbp  = tq >> l2H;
                    const int ih   = ohp * stride + kh - (KS - 1) / 2;
                    const int iw   = owp * stride + kw - (KS - 1) / 2;
                    const bool inb = (KS == 1) ||
                                     ((unsigned)ih < (unsigned)Hin &&
                                      (unsigned)iw < (unsigned)Win);
                    const int ihc = (KS == 1) ? ih : min(max(ih, 0), Hin - 1);
                    const int iwc = (KS == 1) ? iw : min(max(iw, 0), Win - 1);
                    v8h d = *(const v8h*)(src +
                              (((size_t)nbp * Hin + ihc) * Win + iwc) * C + ch);
                    v4i di = __builtin_bit_cast(v4i, d);
                    di &= (inb ? -1 : 0);                // zero padded taps
                    *(v8h*)(As + pixl * rows + ch) = __builtin_bit_cast(v8h, di);
                }
                __syncthreads();
                // ---- K loop: A from LDS, B from global, WMMA ----
                const _Float16* arow = As + col * rows;
                for (int cb = 0; cb < C; cb += 32) {
                    __builtin_prefetch(bp + cb + 256, 0, 1);
                    v8h lo = *(const v8h*)(arow + cb + 16 * khalf);
                    v8h hi = *(const v8h*)(arow + cb + 16 * khalf + 8);
                    v16h a = __builtin_shufflevector(lo, hi,
                                 0, 1, 2, 3, 4, 5, 6, 7,
                                 8, 9, 10, 11, 12, 13, 14, 15);
                    v16h b = *(const v16h*)(bp + cb + 16 * khalf);
                    acc = __builtin_amdgcn_wmma_f32_16x16x32_f16(
                              false, a, false, b, (short)0, acc, false, false);
                }
            }
        }
    }

    float sc, sh;
    if (MODE == 0) {
        const float inv = q0[co] * rsqrtf(q3[co] + 1e-5f);
        sc = inv;
        sh = q1[co] - q2[co] * inv;
    } else {
        sc = 1.0f;
        sh = q0[co];
    }
#pragma unroll
    for (int r = 0; r < 8; ++r) {
        const int pmo = blockIdx.x * 16 + r + 8 * khalf;
        float o = acc[r] * sc + sh;
        if (MODE == 0) o = fmaxf(o, 0.0f);
        out16[(size_t)pmo * Cout + co] = (_Float16)o;        // NHWC, coalesced
        if (FINAL) {
            const int owo = pmo & (Wout - 1);
            const int t2  = pmo >> l2W;
            const int oho = t2 & (Hout - 1);
            const int no  = t2 >> l2H;
            out32[((size_t)(no * Cout + co) * Hout + oho) * Wout + owo] = o;
        }
    }
}

// ---------------------------------------------------------------------------
// NCHW f32 (optional element stride/offset, for ss* time slices) -> NHWC f16
// ---------------------------------------------------------------------------
__global__ void to_nhwc_f16_kernel(const float* __restrict__ src,
                                   _Float16* __restrict__ dst,
                                   int B, int C, int H, int W, int st, int off)
{
    const int idx = blockIdx.x * blockDim.x + threadIdx.x;
    const int total = B * H * W * C;
    if (idx >= total) return;
    const int c = idx % C;
    int p = idx / C;
    const int w = p % W; p /= W;
    const int h = p % H;
    const int n = p / H;
    dst[idx] = (_Float16)src[(((size_t)(n * C + c) * H + h) * W + w)
                             * (size_t)st + off];
}

// ---------------------------------------------------------------------------
// Weight repack: f32 [Cout][Cin][KK] (OIHW) -> f16 [Cout][KK][Cin] (K-contig)
// ---------------------------------------------------------------------------
__global__ void repack_w_kernel(const float* __restrict__ src,
                                _Float16* __restrict__ dst,
                                int Cout, int Cin, int KK)
{
    const int idx = blockIdx.x * blockDim.x + threadIdx.x;
    const int total = Cout * KK * Cin;
    if (idx >= total) return;
    const int ci = idx % Cin;
    int p = idx / Cin;
    const int tap = p % KK;
    const int co  = p / KK;
    dst[idx] = (_Float16)src[((size_t)co * Cin + ci) * KK + tap];
}

// ---------------------------------------------------------------------------
// Depthwise 3x3 + bias on NHWC f16 (memory-bound, plain VALU)
// ---------------------------------------------------------------------------
__global__ void dwconv3x3_nhwc_kernel(const _Float16* __restrict__ x,
                                      const float* __restrict__ w,
                                      const float* __restrict__ bias,
                                      _Float16* __restrict__ out,
                                      int B, int C, int H, int W)
{
    const int idx = blockIdx.x * blockDim.x + threadIdx.x;
    const int total = B * H * W * C;
    if (idx >= total) return;
    const int c = idx % C;
    int p = idx / C;
    const int ww = p % W; p /= W;
    const int hh = p % H;
    const int n  = p / H;
    const float* wp = w + c * 9;
    float s = bias[c];
#pragma unroll
    for (int kh = 0; kh < 3; ++kh) {
        const int ih = hh + kh - 1;
        if ((unsigned)ih >= (unsigned)H) continue;
#pragma unroll
        for (int kw = 0; kw < 3; ++kw) {
            const int iw = ww + kw - 1;
            if ((unsigned)iw >= (unsigned)W) continue;
            s += (float)x[((size_t)(n * H + ih) * W + iw) * C + c] * wp[kh * 3 + kw];
        }
    }
    out[idx] = (_Float16)s;
}

// ---------------------------------------------------------------------------
// LSTM gates on NHWC f16 mix (B,H,W,4*dim); c-state kept f32 for accuracy;
// gathers h into xsel when seq_length[n] == t.
// ---------------------------------------------------------------------------
__global__ void lstm_gates_nhwc_kernel(const _Float16* __restrict__ mix,
                                       _Float16* __restrict__ h,
                                       float* __restrict__ c,
                                       _Float16* __restrict__ xsel,
                                       const int* __restrict__ seqlen, int t,
                                       int B, int dim, int HW)
{
    const int idx = blockIdx.x * blockDim.x + threadIdx.x;
    const int total = B * HW * dim;
    if (idx >= total) return;
    const int ch = idx % dim;
    const int p  = idx / dim;
    const int n  = p / HW;
    const size_t base = (size_t)p * 4 * dim + ch;
    const float f = 1.0f / (1.0f + __expf(-(float)mix[base]));
    const float i = 1.0f / (1.0f + __expf(-(float)mix[base + dim]));
    const float o = 1.0f / (1.0f + __expf(-(float)mix[base + 2 * dim]));
    const float g = tanhf((float)mix[base + 3 * dim]);
    const float ct = f * c[idx] + i * g;
    const float ht = o * tanhf(ct);
    c[idx] = ct;
    h[idx] = (_Float16)ht;
    if (seqlen[n] == t) xsel[idx] = (_Float16)ht;
}

// ---------------------------------------------------------------------------
// Nearest-neighbor 2x upsample, NHWC f16 (H,W are INPUT dims)
// ---------------------------------------------------------------------------
__global__ void upsample2_nhwc_kernel(const _Float16* __restrict__ x,
                                      _Float16* __restrict__ out,
                                      int B, int C, int H, int W)
{
    const int Ho = H * 2, Wo = W * 2;
    const int idx = blockIdx.x * blockDim.x + threadIdx.x;
    const int total = B * Ho * Wo * C;
    if (idx >= total) return;
    const int c = idx % C;
    int p = idx / C;
    const int wo = p % Wo; p /= Wo;
    const int ho = p % Ho;
    const int n  = p / Ho;
    out[idx] = x[((size_t)(n * H + (ho >> 1)) * W + (wo >> 1)) * C + c];
}

// ---------------------------------------------------------------------------
// Host orchestration
// ---------------------------------------------------------------------------
extern "C" void kernel_launch(void* const* d_in, const int* in_sizes, int n_in,
                              void* d_out, int out_size, void* d_ws, size_t ws_size,
                              hipStream_t stream)
{
    (void)in_sizes; (void)n_in; (void)out_size; (void)ws_size;

    auto F = [&](int i) { return (const float*)d_in[i]; };

    const float* f2  = F(0);
    const float* f1  = F(1);
    const float* f0  = F(2);
    const float* ss2 = F(3);
    const float* ss1 = F(4);
    const float* ss0 = F(5);
    const int*   seqlen = (const int*)d_in[6];

    enum { FUSE0 = 7, FUSE1 = 12, FUSE2 = 17,
           LSTM0 = 22, LSTM1 = 26, LSTM2 = 30,
           LAT0 = 34, C3P4 = 39, RED1 = 64, C3P3 = 69,
           BU2 = 94, C3N3 = 99, BU1 = 124, C3N4 = 129 };

    const int B = 4, TS = 6;
    const size_t S0 = (size_t)B * 512 * 16 * 16;   // 524288
    const size_t S1 = (size_t)B * 256 * 32 * 32;   // 1048576
    const size_t S2 = (size_t)B * 128 * 64 * 64;   // 2097152

    char* wsbase = (char*)d_ws;
    size_t off = 0;
    auto alloc  = [&](size_t bytes) {
        off = (off + 63) & ~(size_t)63;
        void* p = wsbase + off; off += bytes; return p;
    };
    auto allocF = [&](size_t n) { return (float*)alloc(n * sizeof(float)); };
    auto allocH = [&](size_t n) { return (_Float16*)alloc(n * sizeof(_Float16)); };

    // Persistent f16 NHWC inputs / state
    _Float16* f0h = allocH(S0); _Float16* f1h = allocH(S1); _Float16* f2h = allocH(S2);
    _Float16* sb0 = allocH(S0); _Float16* sb1 = allocH(S1); _Float16* sb2 = allocH(S2);
    _Float16* h0  = allocH(S0); _Float16* h1  = allocH(S1); _Float16* h2  = allocH(S2);
    float*    c0  = allocF(S0); float*    c1  = allocF(S1); float*    c2  = allocF(S2);
    _Float16* x0h = allocH(S0); _Float16* x1h = allocH(S1); _Float16* x2h = allocH(S2);
    _Float16* pan2h = allocH(S2); _Float16* pan1h = allocH(S1); _Float16* pan0h = allocH(S0);

    // Packed f16 weights
    auto pack = [&](int pb, int Cout, int Cin, int KK) {
        _Float16* dst = allocH((size_t)Cout * Cin * KK);
        const int tot = Cout * Cin * KK;
        repack_w_kernel<<<(tot + 255) / 256, 256, 0, stream>>>(F(pb), dst, Cout, Cin, KK);
        return dst;
    };
    _Float16* Wfuse0 = pack(FUSE0, 512, 1024, 9);
    _Float16* Wfuse1 = pack(FUSE1, 256, 512, 9);
    _Float16* Wfuse2 = pack(FUSE2, 128, 256, 9);
    _Float16* Wpw0   = pack(LSTM0 + 2, 2048, 1024, 1);
    _Float16* Wpw1   = pack(LSTM1 + 2, 1024, 512, 1);
    _Float16* Wpw2   = pack(LSTM2 + 2, 512, 256, 1);
    _Float16* Wlat0  = pack(LAT0, 256, 512, 1);
    _Float16* Wred1  = pack(RED1, 128, 256, 1);
    _Float16* Wbu2   = pack(BU2, 128, 128, 9);
    _Float16* Wbu1   = pack(BU1, 256, 256, 9);
    auto packcsp = [&](int pb, int Cin, int Cout, _Float16** Wc) {
        const int h = Cout / 2;
        Wc[0] = pack(pb,      h,    Cin,   1);
        Wc[1] = pack(pb + 5,  h,    Cin,   1);
        Wc[2] = pack(pb + 10, h,    h,     1);
        Wc[3] = pack(pb + 15, h,    h,     9);
        Wc[4] = pack(pb + 20, Cout, 2 * h, 1);
    };
    _Float16* Wp4[5]; packcsp(C3P4, 512, 256, Wp4);
    _Float16* Wp3[5]; packcsp(C3P3, 256, 128, Wp3);
    _Float16* Wn3[5]; packcsp(C3N3, 256, 256, Wn3);
    _Float16* Wn4[5]; packcsp(C3N4, 512, 512, Wn4);

    // Scratch region (scan scratch; overlaid by FPN-head temps afterwards)
    const size_t scratch = off;
    _Float16* cfh  = allocH(S2);
    _Float16* hhh  = allocH(S2);
    _Float16* mixh = allocH((size_t)B * 512 * 64 * 64);   // 8388608 halves
    off = scratch;
    _Float16* fpn0h  = allocH(S0 / 2);
    _Float16* up0h   = allocH(S1);
    _Float16* fout0h = allocH(S1);
    _Float16* fpn1h  = allocH(S1 / 2);
    _Float16* up1h   = allocH(S2);
    _Float16* t0h    = allocH((size_t)1 << 20);
    _Float16* t1h    = allocH((size_t)1 << 20);
    _Float16* t2h    = allocH((size_t)1 << 20);
    _Float16* p1bh   = allocH(S1 / 2);
    _Float16* p0bh   = allocH(S0 / 2);

    float* out2 = (float*)d_out;   // pan_out2 (4,128,64,64)
    float* out1 = out2 + S2;       // pan_out1 (4,256,32,32)
    float* out0 = out1 + S1;       // pan_out0 (4,512,16,16)

    auto conv = [&](const _Float16* a, int CA, const _Float16* b, int CB,
                    const _Float16* wpk, int pb, int mode,
                    _Float16* o16, float* o32, int Hin, int Cout, int ks, int stride)
    {
        const int Hout = Hin / stride;
        const int M = B * Hout * Hout;
        const int WPB = (Cout >= 128) ? 8 : (Cout / 16);   // waves per block
        dim3 grid(M / 16, Cout / (16 * WPB));
        dim3 blk(32, WPB);
        const float* q0 = F(pb + 1);
        const float* q1 = (mode == 0) ? F(pb + 2) : nullptr;
        const float* q2 = (mode == 0) ? F(pb + 3) : nullptr;
        const float* q3 = (mode == 0) ? F(pb + 4) : nullptr;
        if (ks == 1 && mode == 0 && o32 == nullptr)
            wmma_conv_lds<1, 0, 0><<<grid, blk, 0, stream>>>(
                a, CA, b, CB, wpk, q0, q1, q2, q3, o16, o32,
                Hin, Hin, Hout, Hout, Cout, stride);
        else if (ks == 1 && mode == 0)
            wmma_conv_lds<1, 0, 1><<<grid, blk, 0, stream>>>(
                a, CA, b, CB, wpk, q0, q1, q2, q3, o16, o32,
                Hin, Hin, Hout, Hout, Cout, stride);
        else if (ks == 1)
            wmma_conv_lds<1, 1, 0><<<grid, blk, 0, stream>>>(
                a, CA, b, CB, wpk, q0, q1, q2, q3, o16, o32,
                Hin, Hin, Hout, Hout, Cout, stride);
        else
            wmma_conv_lds<3, 0, 0><<<grid, blk, 0, stream>>>(
                a, CA, b, CB, wpk, q0, q1, q2, q3, o16, o32,
                Hin, Hin, Hout, Hout, Cout, stride);
    };

    auto csp = [&](const _Float16* a, int CA, const _Float16* b, int CB,
                   _Float16** Wc, int pb, _Float16* o16, float* o32, int H, int Cout)
    {
        const int h = Cout / 2;
        conv(a, CA, b, CB,         Wc[0], pb,      0, t0h, nullptr, H, h,    1, 1);
        conv(a, CA, b, CB,         Wc[1], pb + 5,  0, t1h, nullptr, H, h,    1, 1);
        conv(t0h, h, nullptr, 0,   Wc[2], pb + 10, 0, t2h, nullptr, H, h,    1, 1);
        conv(t2h, h, nullptr, 0,   Wc[3], pb + 15, 0, t0h, nullptr, H, h,    3, 1);
        conv(t0h, h, t1h, h,       Wc[4], pb + 20, 0, o16, o32,     H, Cout, 1, 1);
    };

    auto cvt = [&](const float* src, _Float16* dst, int C, int H, int st, int o) {
        const int tot = B * H * H * C;
        to_nhwc_f16_kernel<<<(tot + 255) / 256, 256, 0, stream>>>(src, dst, B, C, H, H, st, o);
    };

    // Input conversion + zero LSTM state
    cvt(f0, f0h, 512, 16, 1, 0);
    cvt(f1, f1h, 256, 32, 1, 0);
    cvt(f2, f2h, 128, 64, 1, 0);
    hipMemsetAsync(h0, 0, S0 * 2, stream); hipMemsetAsync(h1, 0, S1 * 2, stream);
    hipMemsetAsync(h2, 0, S2 * 2, stream); hipMemsetAsync(c0, 0, S0 * 4, stream);
    hipMemsetAsync(c1, 0, S1 * 4, stream); hipMemsetAsync(c2, 0, S2 * 4, stream);

    // ---- ConvLSTM scan ----------------------------------------------------
    for (int t = 0; t < TS; ++t) {
        // scale 0: 16x16, dim 512
        cvt(ss0, sb0, 512, 16, TS, t);
        conv(f0h, 512, sb0, 512, Wfuse0, FUSE0, 0, cfh, nullptr, 16, 512, 3, 1);
        dwconv3x3_nhwc_kernel<<<((int)S0 + 255) / 256, 256, 0, stream>>>(
            h0, F(LSTM0), F(LSTM0 + 1), hhh, B, 512, 16, 16);
        conv(cfh, 512, hhh, 512, Wpw0, LSTM0 + 2, 1, mixh, nullptr, 16, 2048, 1, 1);
        lstm_gates_nhwc_kernel<<<((int)S0 + 255) / 256, 256, 0, stream>>>(
            mixh, h0, c0, x0h, seqlen, t, B, 512, 256);

        // scale 1: 32x32, dim 256
        cvt(ss1, sb1, 256, 32, TS, t);
        conv(f1h, 256, sb1, 256, Wfuse1, FUSE1, 0, cfh, nullptr, 32, 256, 3, 1);
        dwconv3x3_nhwc_kernel<<<((int)S1 + 255) / 256, 256, 0, stream>>>(
            h1, F(LSTM1), F(LSTM1 + 1), hhh, B, 256, 32, 32);
        conv(cfh, 256, hhh, 256, Wpw1, LSTM1 + 2, 1, mixh, nullptr, 32, 1024, 1, 1);
        lstm_gates_nhwc_kernel<<<((int)S1 + 255) / 256, 256, 0, stream>>>(
            mixh, h1, c1, x1h, seqlen, t, B, 256, 1024);

        // scale 2: 64x64, dim 128
        cvt(ss2, sb2, 128, 64, TS, t);
        conv(f2h, 128, sb2, 128, Wfuse2, FUSE2, 0, cfh, nullptr, 64, 128, 3, 1);
        dwconv3x3_nhwc_kernel<<<((int)S2 + 255) / 256, 256, 0, stream>>>(
            h2, F(LSTM2), F(LSTM2 + 1), hhh, B, 128, 64, 64);
        conv(cfh, 128, hhh, 128, Wpw2, LSTM2 + 2, 1, mixh, nullptr, 64, 512, 1, 1);
        lstm_gates_nhwc_kernel<<<((int)S2 + 255) / 256, 256, 0, stream>>>(
            mixh, h2, c2, x2h, seqlen, t, B, 128, 4096);
    }

    // ---- FPN / PAN head ---------------------------------------------------
    conv(x0h, 512, nullptr, 0, Wlat0, LAT0, 0, fpn0h, nullptr, 16, 256, 1, 1);
    upsample2_nhwc_kernel<<<((int)S1 + 255) / 256, 256, 0, stream>>>(fpn0h, up0h, B, 256, 16, 16);
    csp(up0h, 256, x1h, 256, Wp4, C3P4, fout0h, nullptr, 32, 256);

    conv(fout0h, 256, nullptr, 0, Wred1, RED1, 0, fpn1h, nullptr, 32, 128, 1, 1);
    upsample2_nhwc_kernel<<<((int)S2 + 255) / 256, 256, 0, stream>>>(fpn1h, up1h, B, 128, 32, 32);
    csp(up1h, 128, x2h, 128, Wp3, C3P3, pan2h, out2, 64, 128);       // pan_out2

    conv(pan2h, 128, nullptr, 0, Wbu2, BU2, 0, p1bh, nullptr, 64, 128, 3, 2);
    csp(p1bh, 128, fpn1h, 128, Wn3, C3N3, pan1h, out1, 32, 256);     // pan_out1

    conv(pan1h, 256, nullptr, 0, Wbu1, BU1, 0, p0bh, nullptr, 32, 256, 3, 2);
    csp(p0bh, 256, fpn0h, 256, Wn4, C3N4, pan0h, out0, 16, 512);     // pan_out0
}
